// Conv2DExperimental_71829033058621
// MI455X (gfx1250) — matrix operand order repaired and checked by reading, not compiled, not run on hardware
//
#include <hip/hip_runtime.h>
#include <stdint.h>

typedef __attribute__((ext_vector_type(16))) __bf16        v16bf;
typedef __attribute__((ext_vector_type(8)))  float         v8f;
typedef __attribute__((ext_vector_type(4)))  unsigned int  uint4v;

#define HH 224
#define WW 224
#define CIN 64
#define COUT 64
#define NBATCH 32
#define KTOT 576            // 9 kernel taps * 64 input channels
#define TILE_W 32           // pixels per block
#define XCOLS 34            // 32 + 2 halo
#define XROWS 3
#define XSTRIDE (XROWS * XCOLS)          // 102 dwords per channel-pair
#define XPK_DWORDS (32 * XSTRIDE)        // 3264 dwords = 13056 B LDS

__device__ __forceinline__ float softplusf(float x) {
    return (x > 15.0f) ? x : log1pf(expf(x));
}

// f32 -> bf16 round-to-nearest-even
__device__ __forceinline__ unsigned short f2bf(float f) {
    unsigned int u = __float_as_uint(f);
    u += 0x7FFFu + ((u >> 16) & 1u);
    return (unsigned short)(u >> 16);
}

// ---------------------------------------------------------------------------
// Kernel 1: reparameterized weight/bias sampling.
//   w[o][i][a] = loc[o][i][a] + sum_b L[a][b] * eps[b],  L = tril(-1) + softplus(diag)
// Stored bf16 in GEMM-A layout: WA[o][k], k = a*64 + i  (tap-major, channel-minor)
// ---------------------------------------------------------------------------
__global__ void sample_weights_kernel(const float* __restrict__ wloc,
                                      const float* __restrict__ wL,
                                      const float* __restrict__ bloc,
                                      const float* __restrict__ bro,
                                      const float* __restrict__ epsw,
                                      const float* __restrict__ epsb,
                                      unsigned short* __restrict__ WA,
                                      float* __restrict__ bout)
{
    int t = blockIdx.x * blockDim.x + threadIdx.x;
    if (t < COUT) bout[t] = bloc[t] + epsb[t] * softplusf(bro[t]);
    if (t >= COUT * CIN) return;

    int o = t >> 6, i = t & 63;
    const float* Lb = wL   + (size_t)t * 81;
    const float* eb = epsw + (size_t)t * 9;

    float e[9];
#pragma unroll
    for (int b = 0; b < 9; ++b) e[b] = eb[b];

#pragma unroll
    for (int a = 0; a < 9; ++a) {
        float acc = wloc[(size_t)t * 9 + a];
#pragma unroll
        for (int b = 0; b < 9; ++b) {
            if (b < a)       acc += Lb[a * 9 + b] * e[b];
            else if (b == a) acc += softplusf(Lb[a * 9 + a]) * e[b];
        }
        WA[(size_t)o * KTOT + a * 64 + i] = f2bf(acc);
    }
}

// ---------------------------------------------------------------------------
// Kernel 2: implicit-GEMM 3x3 SAME conv via v_wmma_f32_16x16x32_bf16.
// Block = 128 threads (4 waves), computes 64 out-channels x 32 pixels of one
// (n, oy) row. Wave wv handles channels [16*wv, 16*wv+16) x 32 pixels.
// ---------------------------------------------------------------------------
__global__ __launch_bounds__(128) void conv_wmma_kernel(
    const float* __restrict__ x,
    const unsigned short* __restrict__ WA,
    const float* __restrict__ biasf,
    float* __restrict__ out)
{
    __shared__ unsigned int xpk[XPK_DWORDS];   // [cpair][row][col] packed bf16x2

    const int tid = threadIdx.x;
    const int blk = blockIdx.x;
    const int tx  = blk % 7;
    const int oy  = (blk / 7) % HH;
    const int n   = blk / (7 * HH);
    const int ox0 = tx * TILE_W;

    // ---- Stage x halo tile: 64 ch x 3 rows x 34 cols, bf16 channel-pairs ----
    for (int tc = tid; tc < XPK_DWORDS; tc += 128) {
        int cp  = tc / XSTRIDE;
        int rem = tc % XSTRIDE;
        int r   = rem / XCOLS;
        int cc  = rem % XCOLS;
        int iy  = oy - 1 + r;
        int ix  = ox0 - 1 + cc;
        unsigned int pk = 0u;
        if ((unsigned)iy < (unsigned)HH && (unsigned)ix < (unsigned)WW) {
            size_t base = ((size_t)(n * CIN + cp * 2) * HH + iy) * WW + ix;
            float f0 = x[base];
            float f1 = x[base + (size_t)HH * WW];
            pk = (unsigned int)f2bf(f0) | ((unsigned int)f2bf(f1) << 16);
        }
        xpk[tc] = pk;
    }
    __syncthreads();

    const int lane = tid & 31;
    const int wv   = tid >> 5;
    const int m0   = wv * 16;       // output-channel tile base
    const int lrow = lane & 15;
    const int lhi  = lane >> 4;

    v8f acc0 = {};                  // pixels ox0 + [0,16)
    v8f acc1 = {};                  // pixels ox0 + [16,32)

#pragma unroll
    for (int kb = 0; kb < 18; ++kb) {
        const int a  = kb >> 1;             // kernel tap 0..8
        const int ky = a / 3;
        const int kx = a % 3;
        const int i0 = (kb & 1) * 32;       // input-channel block base

        // A fragment (16x32 bf16): lane row M = m0+lrow; K = lhi*16 + [0,16)
        union { uint4v q[2]; v16bf v; } af;
        const int aidx = (m0 + lrow) * KTOT + kb * 32 + lhi * 16;
        af.q[0] = *(const uint4v*)(WA + aidx);
        af.q[1] = *(const uint4v*)(WA + aidx + 8);

        // B fragments (32x16 bf16): lane col N = lrow; K = lhi*16 + [0,16)
        // dword j holds channel pair (i0 + lhi*16 + 2j, +1) at tap (ky,kx)
        const int cp0   = (i0 + lhi * 16) >> 1;
        const int bbase = cp0 * XSTRIDE + ky * XCOLS + lrow + kx;
        union { unsigned int u[8]; v16bf v; } bf0, bf1;
#pragma unroll
        for (int j = 0; j < 8; ++j) {
            bf0.u[j] = xpk[bbase +      j * XSTRIDE];
            bf1.u[j] = xpk[bbase + 16 + j * XSTRIDE];
        }

        acc0 = __builtin_amdgcn_wmma_f32_16x16x32_bf16(
                   false, af.v, false, bf0.v, (short)0, acc0, false, false);
        acc1 = __builtin_amdgcn_wmma_f32_16x16x32_bf16(
                   false, af.v, false, bf1.v, (short)0, acc1, false, false);
    }

    // ---- Epilogue: bias add + store. C/D layout: VGPR v -> M = v + 8*lhi, N = lrow
#pragma unroll
    for (int v = 0; v < 8; ++v) {
        int m = m0 + v + lhi * 8;
        float bv = biasf[m];
        size_t o0 = ((size_t)(n * COUT + m) * HH + oy) * WW + (ox0 + lrow);
        out[o0]      = acc0[v] + bv;
        out[o0 + 16] = acc1[v] + bv;
    }
}

// ---------------------------------------------------------------------------
extern "C" void kernel_launch(void* const* d_in, const int* in_sizes, int n_in,
                              void* d_out, int out_size, void* d_ws, size_t ws_size,
                              hipStream_t stream)
{
    const float* x    = (const float*)d_in[0];
    const float* wloc = (const float*)d_in[1];
    const float* wL   = (const float*)d_in[2];
    const float* bloc = (const float*)d_in[3];
    const float* bro  = (const float*)d_in[4];
    const float* epsw = (const float*)d_in[5];
    const float* epsb = (const float*)d_in[6];
    float* out = (float*)d_out;

    // ws layout: bf16 weights [64][576] (73728 B), then f32 bias[64]
    unsigned short* WA    = (unsigned short*)d_ws;
    float*          biasf = (float*)((char*)d_ws + (size_t)COUT * KTOT * sizeof(unsigned short));

    sample_weights_kernel<<<16, 256, 0, stream>>>(wloc, wL, bloc, bro, epsw, epsb, WA, biasf);

    const int nblocks = NBATCH * HH * (WW / TILE_W);   // 32*224*7 = 50176
    conv_wmma_kernel<<<nblocks, 128, 0, stream>>>(x, WA, biasf, out);
}
